// DynamicAttentionModulation_56659208569094
// MI455X (gfx1250) — matrix operand order repaired and checked by reading, not compiled, not run on hardware
//
#include <hip/hip_runtime.h>
#include <hip/hip_bf16.h>

// ---------------- problem constants ----------------
#define BATCH 16
#define SEQ   1024
#define DDIM  1024
#define UDIM  1024
#define MROWS (BATCH * SEQ)   // 16384 global rows

typedef __attribute__((ext_vector_type(16))) __bf16 v16bf;
typedef __attribute__((ext_vector_type(8)))  __bf16 v8bf;
typedef __attribute__((ext_vector_type(8)))  float  v8f;

static __device__ __forceinline__ v16bf load_a_frag(const __bf16* row, int k0, int koffA)
{
    // A frag, this code's contiguous k order (B same): lane<16 halves0-15 = K 0..15
    //                           lane>=16 halves0-15 = K 16..31.
    v8bf a0 = *(const v8bf*)(row + k0 + koffA);
    v8bf a1 = *(const v8bf*)(row + k0 + koffA + 8);
    v16bf a;
    #pragma unroll
    for (int i = 0; i < 8; ++i) { a[i] = a0[i]; a[8 + i] = a1[i]; }
    return a;
}

// =====================================================================
// Kernel 0a: x fp32 -> bf16 (one-time, keeps cvt out of GEMM loops)
// =====================================================================
__global__ __launch_bounds__(256) void k_cvt_x(const float* __restrict__ x,
                                               __bf16* __restrict__ xb)
{
    size_t i = ((size_t)blockIdx.x * 256 + threadIdx.x) * 8;
    v8f v = *(const v8f*)(x + i);
    v8bf o;
    #pragma unroll
    for (int j = 0; j < 8; ++j) o[j] = (__bf16)v[j];
    *(v8bf*)(xb + i) = o;
}

// =====================================================================
// Kernel 0b: weight transpose+convert  W[D,U] fp32 -> Wt[U,D] bf16.
//   32x32 LDS tile transpose; coalesced global reads AND writes.
//   With Wt, the GEMM B-operand (per-lane column n, K-contiguous) is a
//   single contiguous 32-byte load instead of 16 strided fp32 loads.
// =====================================================================
__global__ __launch_bounds__(256) void k_wt_transpose(
    const float* __restrict__ Wq, const float* __restrict__ Wk,
    const float* __restrict__ Wv, const float* __restrict__ Wm,
    __bf16* __restrict__ Wt)
{
    __shared__ float tile[32][33];
    const int which = blockIdx.z;
    const float* W = (which == 0) ? Wq : (which == 1) ? Wk : (which == 2) ? Wv : Wm;
    __bf16* out = Wt + (size_t)which * UDIM * DDIM;

    const int u0 = blockIdx.x * 32;
    const int d0 = blockIdx.y * 32;
    const int c  = threadIdx.x & 31;
    const int r4 = threadIdx.x >> 5;          // 0..7, 4 rows each

    #pragma unroll
    for (int rr = 0; rr < 4; ++rr) {
        int row = r4 * 4 + rr;
        tile[row][c] = W[(size_t)(d0 + row) * UDIM + u0 + c];
    }
    __syncthreads();
    #pragma unroll
    for (int rr = 0; rr < 4; ++rr) {
        int urow = r4 * 4 + rr;
        out[(size_t)(u0 + urow) * DDIM + d0 + c] = (__bf16)tile[c][urow];
    }
}

// =====================================================================
// Kernel 1: fused QKV+Mod projections, 64x64 tile per wave.
//   16 f32 accumulators (128 VGPRs), 4 A-frags + 4 B-frags per k-step,
//   16 WMMAs per k-step -> ~32 flops/byte of fragment traffic.
//   blockIdx.z selects which weight / output.
// =====================================================================
__global__ __launch_bounds__(32) void k_qkvm_proj(
    const __bf16* __restrict__ xb, const __bf16* __restrict__ Wt,
    const float* __restrict__ bm,
    __bf16* __restrict__ Q, __bf16* __restrict__ Kb,
    __bf16* __restrict__ Vt, __bf16* __restrict__ Md)
{
    const int lane  = threadIdx.x;
    const int n0    = blockIdx.x * 64;
    const int m0    = blockIdx.y * 64;
    const int which = blockIdx.z;
    const __bf16* W = Wt + (size_t)which * UDIM * DDIM;   // [U, D] bf16

    const int koffA = (lane >> 4) * 16;
    const int koffB = (lane >> 4) * 16;
    const int lm    = lane & 15;

    const __bf16* arow[4];
    const __bf16* brow[4];
    #pragma unroll
    for (int i = 0; i < 4; ++i) arow[i] = xb + (size_t)(m0 + i * 16 + lm) * DDIM;
    #pragma unroll
    for (int j = 0; j < 4; ++j) brow[j] = W  + (size_t)(n0 + j * 16 + lm) * DDIM;

    v8f acc[4][4] = {};
    for (int k0 = 0; k0 < DDIM; k0 += 32) {
        v16bf a[4], b[4];
        #pragma unroll
        for (int j = 0; j < 4; ++j) {
            b[j] = *(const v16bf*)(brow[j] + k0 + koffB);
            __builtin_prefetch(brow[j] + k0 + 32 + koffB, 0, 1);
        }
        #pragma unroll
        for (int i = 0; i < 4; ++i) a[i] = load_a_frag(arow[i], k0, koffA);
        #pragma unroll
        for (int i = 0; i < 4; ++i)
            #pragma unroll
            for (int j = 0; j < 4; ++j)
                acc[i][j] = __builtin_amdgcn_wmma_f32_16x16x32_bf16(
                                false, a[i], false, b[j], (short)0, acc[i][j], false, false);
    }

    // D layout: VGPR r -> M = r + 8*(lane>=16), N = lane&15
    const int N     = lane & 15;
    const int Mbase = (lane >> 4) * 8;

    if (which == 0 || which == 1) {
        __bf16* out = (which == 0) ? Q : Kb;
        #pragma unroll
        for (int i = 0; i < 4; ++i)
            #pragma unroll
            for (int j = 0; j < 4; ++j)
                #pragma unroll
                for (int r = 0; r < 8; ++r) {
                    int m = m0 + i * 16 + Mbase + r;
                    out[(size_t)m * UDIM + n0 + j * 16 + N] = (__bf16)acc[i][j][r];
                }
    } else if (which == 2) {
        #pragma unroll
        for (int i = 0; i < 4; ++i)
            #pragma unroll
            for (int j = 0; j < 4; ++j)
                #pragma unroll
                for (int r = 0; r < 8; ++r) {
                    int m    = m0 + i * 16 + Mbase + r;
                    int bidx = m >> 10;
                    int t    = m & (SEQ - 1);
                    Vt[((size_t)bidx * UDIM + (n0 + j * 16 + N)) * SEQ + t] = (__bf16)acc[i][j][r];
                }
    } else {
        #pragma unroll
        for (int j = 0; j < 4; ++j) {
            float bias = bm[n0 + j * 16 + N];
            #pragma unroll
            for (int i = 0; i < 4; ++i)
                #pragma unroll
                for (int r = 0; r < 8; ++r) {
                    int m = m0 + i * 16 + Mbase + r;
                    float s = 1.0f / (1.0f + __expf(-(acc[i][j][r] + bias)));
                    Md[(size_t)m * UDIM + n0 + j * 16 + N] = (__bf16)s;
                }
        }
    }
}

// =====================================================================
// Kernel 2: scores = Q K^T / 32, softmax over t, gate by Mod -> P bf16.
//   8 waves per 16-row strip; each wave computes 4 score tiles per
//   A-fragment load (A reused 4x), scores staged in 64 KB LDS.
//   Softmax: 16 threads/row, width-16 __shfl_xor reductions (wave32).
// =====================================================================
__global__ __launch_bounds__(256) void k_attn_softmax(
    const __bf16* __restrict__ Q, const __bf16* __restrict__ Kb,
    const __bf16* __restrict__ Md, __bf16* __restrict__ P)
{
    __shared__ float sc[16 * SEQ];            // 64 KB of the 320 KB WGP LDS

    const int tid  = threadIdx.x;
    const int lane = tid & 31;
    const int wave = tid >> 5;                // 0..7
    const int gm   = blockIdx.x;              // 16-row strip over B*S
    const size_t row0 = (size_t)gm * 16;
    const int bidx = gm >> 6;

    const int koffA = (lane >> 4) * 16;
    const int koffB = (lane >> 4) * 16;
    const int N     = lane & 15;
    const int Mbase = (lane >> 4) * 8;

    const __bf16* qrow = Q  + (row0 + (lane & 15)) * UDIM;
    const __bf16* Kbat = Kb + (size_t)bidx * SEQ * UDIM;
    const float scale  = 0.03125f;            // 1/sqrt(1024)

    #pragma unroll
    for (int g = 0; g < 2; ++g) {
        const int t0 = (wave * 2 + g) * 64;   // this wave's 4 tiles (64 cols)
        const __bf16* krow[4];
        #pragma unroll
        for (int j = 0; j < 4; ++j)
            krow[j] = Kbat + (size_t)(t0 + j * 16 + (lane & 15)) * UDIM;

        v8f acc[4] = {};
        for (int k0 = 0; k0 < UDIM; k0 += 32) {
            v16bf a = load_a_frag(qrow, k0, koffA);
            #pragma unroll
            for (int j = 0; j < 4; ++j) {
                v16bf b = *(const v16bf*)(krow[j] + k0 + koffB);
                acc[j] = __builtin_amdgcn_wmma_f32_16x16x32_bf16(
                             false, a, false, b, (short)0, acc[j], false, false);
            }
        }
        #pragma unroll
        for (int j = 0; j < 4; ++j)
            #pragma unroll
            for (int r = 0; r < 8; ++r)
                sc[(Mbase + r) * SEQ + t0 + j * 16 + N] = acc[j][r] * scale;
    }
    __syncthreads();

    // softmax: 16 threads per row, 64 columns per thread
    const int row = tid >> 4;
    const int seg = tid & 15;
    float* srow = sc + row * SEQ;

    float mx = -3.0e38f;
    for (int c = seg * 64; c < seg * 64 + 64; ++c) mx = fmaxf(mx, srow[c]);
    #pragma unroll
    for (int off = 1; off < 16; off <<= 1) mx = fmaxf(mx, __shfl_xor(mx, off, 16));

    float sum = 0.0f;
    for (int c = seg * 64; c < seg * 64 + 64; ++c) {
        float e = __expf(srow[c] - mx);
        srow[c] = e;
        sum += e;
    }
    #pragma unroll
    for (int off = 1; off < 16; off <<= 1) sum += __shfl_xor(sum, off, 16);
    const float inv = 1.0f / sum;

    const __bf16* mrow = Md + (row0 + row) * UDIM;
    __bf16*       prow = P  + (row0 + row) * SEQ;
    for (int c = seg * 64; c < seg * 64 + 64; ++c)
        prow[c] = (__bf16)(srow[c] * inv * (float)mrow[c]);
}

// =====================================================================
// Kernel 3: context = P @ V per batch, 64x64 tile per wave.
//   A = P row-major (contiguous), B = Vt[B,U,S] (contiguous 32B loads).
// =====================================================================
__global__ __launch_bounds__(32) void k_context(
    const __bf16* __restrict__ P, const __bf16* __restrict__ Vt,
    float* __restrict__ out)
{
    const int lane = threadIdx.x;
    const int n0   = blockIdx.x * 64;
    const int m0   = blockIdx.y * 64;
    const int bidx = m0 >> 10;                // 64 | 1024 so tiles never straddle batches

    const int koffA = (lane >> 4) * 16;
    const int koffB = (lane >> 4) * 16;
    const int lm    = lane & 15;

    const __bf16* arow[4];
    const __bf16* brow[4];
    #pragma unroll
    for (int i = 0; i < 4; ++i) arow[i] = P  + (size_t)(m0 + i * 16 + lm) * SEQ;
    #pragma unroll
    for (int j = 0; j < 4; ++j)
        brow[j] = Vt + ((size_t)bidx * UDIM + (n0 + j * 16 + lm)) * SEQ;

    v8f acc[4][4] = {};
    for (int k0 = 0; k0 < SEQ; k0 += 32) {
        v16bf a[4], b[4];
        #pragma unroll
        for (int j = 0; j < 4; ++j) {
            b[j] = *(const v16bf*)(brow[j] + k0 + koffB);
            __builtin_prefetch(brow[j] + k0 + 32 + koffB, 0, 1);
        }
        #pragma unroll
        for (int i = 0; i < 4; ++i) a[i] = load_a_frag(arow[i], k0, koffA);
        #pragma unroll
        for (int i = 0; i < 4; ++i)
            #pragma unroll
            for (int j = 0; j < 4; ++j)
                acc[i][j] = __builtin_amdgcn_wmma_f32_16x16x32_bf16(
                                false, a[i], false, b[j], (short)0, acc[i][j], false, false);
    }

    const int N     = lane & 15;
    const int Mbase = (lane >> 4) * 8;
    #pragma unroll
    for (int i = 0; i < 4; ++i)
        #pragma unroll
        for (int j = 0; j < 4; ++j)
            #pragma unroll
            for (int r = 0; r < 8; ++r)
                out[(size_t)(m0 + i * 16 + Mbase + r) * UDIM + n0 + j * 16 + N] = acc[i][j][r];
}

// =====================================================================
// launcher
// =====================================================================
extern "C" void kernel_launch(void* const* d_in, const int* in_sizes, int n_in,
                              void* d_out, int out_size, void* d_ws, size_t ws_size,
                              hipStream_t stream)
{
    const float* x  = (const float*)d_in[0];
    const float* Wq = (const float*)d_in[1];
    const float* Wk = (const float*)d_in[2];
    const float* Wv = (const float*)d_in[3];
    const float* Wm = (const float*)d_in[4];
    const float* bm = (const float*)d_in[5];

    // workspace layout (bf16): xb | Q | K | Vt | Md | P | Wt(x4)
    constexpr size_t SZ  = (size_t)MROWS * UDIM * sizeof(__bf16);  // 32 MB
    constexpr size_t WSZ = (size_t)UDIM * DDIM * sizeof(__bf16);   //  2 MB
    char* ws = (char*)d_ws;
    __bf16* xb = (__bf16*)(ws + 0 * SZ);
    __bf16* Q  = (__bf16*)(ws + 1 * SZ);
    __bf16* Kb = (__bf16*)(ws + 2 * SZ);
    __bf16* Vt = (__bf16*)(ws + 3 * SZ);
    __bf16* Md = (__bf16*)(ws + 4 * SZ);
    __bf16* P  = (__bf16*)(ws + 5 * SZ);
    __bf16* Wt = (__bf16*)(ws + 6 * SZ);      // 4 transposed bf16 weights

    k_cvt_x<<<dim3((MROWS * (size_t)DDIM) / (256 * 8)), 256, 0, stream>>>(x, xb);
    k_wt_transpose<<<dim3(UDIM / 32, DDIM / 32, 4), 256, 0, stream>>>(Wq, Wk, Wv, Wm, Wt);

    k_qkvm_proj<<<dim3(UDIM / 64, MROWS / 64, 4), 32, 0, stream>>>(xb, Wt, bm, Q, Kb, Vt, Md);
    k_attn_softmax<<<dim3(MROWS / 16), 256, 0, stream>>>(Q, Kb, Md, P);
    k_context<<<dim3(UDIM / 64, MROWS / 64), 32, 0, stream>>>(P, Vt, (float*)d_out);
}
